// UnifiedLIV_16595753632149
// MI455X (gfx1250) — matrix operand order repaired and checked by reading, not compiled, not run on hardware
//
#include <hip/hip_runtime.h>
#include <hip/hip_bf16.h>

#define DIMX 512
#define HEADS 8
#define SEQ 2048
#define BATCH 2
#define HD 64
#define NTOK (BATCH * SEQ)   // 4096
#define NBH (BATCH * HEADS)  // 16

typedef __bf16 v16bf __attribute__((ext_vector_type(16)));
typedef float  v8f  __attribute__((ext_vector_type(8)));

union TileU { v16bf v; uint4 q[2]; unsigned short u[16]; };

__device__ __forceinline__ unsigned short f2bf(float f) {
    unsigned int u = __float_as_uint(f);
    u += 0x7FFFu + ((u >> 16) & 1u);     // round-to-nearest-even
    return (unsigned short)(u >> 16);
}

__device__ __forceinline__ v8f vzero() {
    v8f z;
#pragma unroll
    for (int i = 0; i < 8; ++i) z[i] = 0.0f;
    return z;
}

// A-operand (16xK=32 bf16): lane<16 holds K {0..7,16..23}; lane>=16 {8..15,24..31}.
// Caller passes row already resolved (= tile_row_base + (lane&15)).
__device__ __forceinline__ v16bf load_a(const unsigned short* base, int ld, int row, int kbase) {
    int lane = threadIdx.x & 31;
    int halfk = ((lane >> 4) & 1) << 3;          // 0 or 8
    const unsigned short* p = base + row * ld + kbase + halfk;
    TileU t;
    t.q[0] = *(const uint4*)(p);
    t.q[1] = *(const uint4*)(p + 16);
    return t.v;
}

// B-operand (K=32 x 16 bf16): lanes 0-15 hold K {0..7,16..23}, lanes 16-31 hold K {8..15,24..31}
// (same K order as load_a). Caller passes row = N index (= nbase + (lane&15)).
__device__ __forceinline__ v16bf load_b(const unsigned short* base, int ld, int row, int kbase) {
    int lane = threadIdx.x & 31;
    int kh = ((lane >> 4) & 1) << 3;             // 0 or 8
    const unsigned short* p = base + row * ld + kbase + kh;
    TileU t;
    t.q[0] = *(const uint4*)(p);
    t.q[1] = *(const uint4*)(p + 16);
    return t.v;
}

__device__ __forceinline__ v8f wmma_bf16(v16bf a, v16bf b, v8f c) {
    return __builtin_amdgcn_wmma_f32_16x16x32_bf16(false, a, false, b, (short)0, c, false, false);
}

// ---------------------------------------------------------------- fp32->bf16
__global__ void cvt_bf16_kernel(const float* __restrict__ src,
                                unsigned short* __restrict__ dst, int n) {
    int i = blockIdx.x * 256 + threadIdx.x;
    if (i < n) dst[i] = f2bf(src[i]);
}

// ---------------------------------------------------------- projection GEMMs
// out[4096,512] = xb[4096,512] @ W[p]^T + bias ; p==2 applies sigmoid(.+A_log)
__global__ void proj_kernel(const unsigned short* __restrict__ xb,
                            const unsigned short* __restrict__ wb4,
                            const float* __restrict__ biasB, const float* __restrict__ biasC,
                            const float* __restrict__ biasS, const float* __restrict__ biasV,
                            const float* __restrict__ A_log,
                            float* __restrict__ Bm, float* __restrict__ Cm,
                            float* __restrict__ Am, float* __restrict__ Vm) {
    int p = blockIdx.z;
    int lane = threadIdx.x & 31;
    int w = threadIdx.x >> 5;
    int lr = lane & 15;
    int mhi = (lane >> 4) * 8;
    int mbase = blockIdx.x * 64 + w * 16;
    int nbase = blockIdx.y * 64;
    const unsigned short* W = wb4 + (size_t)p * DIMX * DIMX;

    v8f acc[4];
#pragma unroll
    for (int t = 0; t < 4; ++t) acc[t] = vzero();

    int arow = mbase + lr;
    for (int k = 0; k < DIMX; k += 32) {
        v16bf a = load_a(xb, DIMX, arow, k);
#pragma unroll
        for (int t = 0; t < 4; ++t) {
            v16bf b = load_b(W, DIMX, nbase + t * 16 + lr, k);
            acc[t] = wmma_bf16(a, b, acc[t]);
        }
    }

    const float* bias = (p == 0) ? biasB : (p == 1) ? biasC : (p == 2) ? biasS : biasV;
    float* out = (p == 0) ? Bm : (p == 1) ? Cm : (p == 2) ? Am : Vm;
#pragma unroll
    for (int t = 0; t < 4; ++t) {
        int n = nbase + t * 16 + lr;
        float bv = bias[n];
        float al = (p == 2) ? A_log[n] : 0.0f;
#pragma unroll
        for (int r = 0; r < 8; ++r) {
            int m = mbase + mhi + r;
            float v = acc[t][r] + bv;
            if (p == 2) v = 1.0f / (1.0f + __expf(-(v + al)));
            out[(size_t)m * DIMX + n] = v;
        }
    }
}

// ----------------------------------------------- sequential gate scan (fp32)
// One block per (b,h); thread c owns channel c. Produces CG=C*G, BH=B/H (bf16),
// Vt = V transposed to [d,L] (bf16), Diag = C.B (fp32).
__global__ void scan_kernel(const float* __restrict__ Bm, const float* __restrict__ Cm,
                            const float* __restrict__ Vm, const float* __restrict__ Am,
                            unsigned short* __restrict__ CG, unsigned short* __restrict__ BHm,
                            unsigned short* __restrict__ Vt, float* __restrict__ Diag) {
    int bh = blockIdx.x;
    int b = bh >> 3, h = bh & 7;
    int c = threadIdx.x;                 // 0..63
    __shared__ float red[64];
    int col = h * HD + c;
    float cum = 1.0f;                    // cumA[l-1] = prod_{k=1..l-1} A_k
    for (int l = 0; l < SEQ; ++l) {
        size_t gi = ((size_t)(b * SEQ) + l) * DIMX + col;
        float a = Am[gi], bb = Bm[gi], cc = Cm[gi], vv = Vm[gi];
        float G = cum;
        float Hd = (l == 0) ? 1.0f : (cum + 1e-8f);
        size_t oi = ((size_t)bh * SEQ + l) * HD + c;
        CG[oi] = f2bf(cc * G);
        BHm[oi] = f2bf(bb / Hd);
        Vt[((size_t)bh * HD + c) * SEQ + l] = f2bf(vv);
        red[c] = cc * bb;
        __syncthreads();
#pragma unroll
        for (int s = 32; s > 0; s >>= 1) {
            if (c < s) red[c] += red[c + s];
            __syncthreads();
        }
        if (c == 0) Diag[(size_t)bh * SEQ + l] = red[0];
        __syncthreads();
        if (l > 0) cum *= a;
    }
}

// --------------------------------------------------- causal mixing + out-mix
// grid (SEQ/64, NBH), block 128. Wave owns 16 query rows x full d=64.
__global__ void attn_kernel(const unsigned short* __restrict__ CG,
                            const unsigned short* __restrict__ BHm,
                            const unsigned short* __restrict__ Vt,
                            const float* __restrict__ Diag,
                            const unsigned short* __restrict__ Wgb,
                            float* __restrict__ Y) {
    __shared__ __align__(16) unsigned short ldsT[4][16 * 32];
    __shared__ __align__(16) unsigned short ldsM[4][16 * 64];
    int bh = blockIdx.y;
    int b = bh >> 3, h = bh & 7;
    int lane = threadIdx.x & 31;
    int w = threadIdx.x >> 5;
    int lr = lane & 15;
    int mhi = (lane >> 4) * 8;
    int qbase = blockIdx.x * 64 + w * 16;

    const unsigned short* cg = CG + (size_t)bh * SEQ * HD;
    const unsigned short* bp = BHm + (size_t)bh * SEQ * HD;
    const unsigned short* vt = Vt + (size_t)bh * HD * SEQ;
    const float* dg = Diag + (size_t)bh * SEQ;
    unsigned short* Tt = ldsT[w];
    unsigned short* Mx = ldsM[w];

    // query tile of CG: A-operand, K = 64 channels in two 32-chunks
    v16bf aCG0 = load_a(cg, HD, qbase + lr, 0);
    v16bf aCG1 = load_a(cg, HD, qbase + lr, 32);

    v8f macc[4];
#pragma unroll
    for (int t = 0; t < 4; ++t) macc[t] = vzero();

    for (int kb = 0; kb < qbase + 16; kb += 32) {
#pragma unroll
        for (int s = 0; s < 2; ++s) {
            int kjn = kb + s * 16 + lr;                  // key index of this lane-column
            int krow = (kjn < SEQ) ? kjn : (SEQ - 1);    // clamp OOB (masked below)
            v8f tacc = vzero();
            v16bf b0 = load_b(bp, HD, krow, 0);
            tacc = wmma_bf16(aCG0, b0, tacc);
            v16bf b1 = load_b(bp, HD, krow, 32);
            tacc = wmma_bf16(aCG1, b1, tacc);
#pragma unroll
            for (int r = 0; r < 8; ++r) {
                int qi = qbase + mhi + r;
                float t = tacc[r];
                t = (qi > kjn) ? t : ((qi == kjn) ? dg[qi] : 0.0f);
                Tt[(mhi + r) * 32 + s * 16 + lr] = f2bf(t);
            }
        }
        // same-wave DS ops are in-order; wait + compiler barrier before re-read
        asm volatile("s_wait_dscnt 0x0" ::: "memory");
        v16bf aT = load_a(Tt, 32, lr, 0);
#pragma unroll
        for (int t = 0; t < 4; ++t) {
            v16bf bv = load_b(vt, SEQ, t * 16 + lr, kb); // Vt[d, l]: contiguous along K
            macc[t] = wmma_bf16(aT, bv, macc[t]);
        }
        asm volatile("s_wait_dscnt 0x0" ::: "memory");
    }

    // per-head 64x64 output mix: Y = mixed @ Wg[h]^T
#pragma unroll
    for (int t = 0; t < 4; ++t)
#pragma unroll
        for (int r = 0; r < 8; ++r)
            Mx[(mhi + r) * 64 + t * 16 + lr] = f2bf(macc[t][r]);
    asm volatile("s_wait_dscnt 0x0" ::: "memory");

    v16bf aM0 = load_a(Mx, 64, lr, 0);
    v16bf aM1 = load_a(Mx, 64, lr, 32);
    const unsigned short* wg = Wgb + (size_t)h * HD * HD;
#pragma unroll
    for (int t = 0; t < 4; ++t) {
        v8f y = vzero();
        v16bf bw0 = load_b(wg, HD, t * 16 + lr, 0);
        y = wmma_bf16(aM0, bw0, y);
        v16bf bw1 = load_b(wg, HD, t * 16 + lr, 32);
        y = wmma_bf16(aM1, bw1, y);
        int e = t * 16 + lr;
#pragma unroll
        for (int r = 0; r < 8; ++r) {
            int l = qbase + mhi + r;
            Y[((size_t)(b * SEQ) + l) * DIMX + h * HD + e] = y[r];
        }
    }
}

extern "C" void kernel_launch(void* const* d_in, const int* in_sizes, int n_in,
                              void* d_out, int out_size, void* d_ws, size_t ws_size,
                              hipStream_t stream) {
    const float* x    = (const float*)d_in[0];
    const float* WBw  = (const float*)d_in[1];
    const float* WBb  = (const float*)d_in[2];
    const float* WCw  = (const float*)d_in[3];
    const float* WCb  = (const float*)d_in[4];
    const float* WSw  = (const float*)d_in[5];
    const float* WSb  = (const float*)d_in[6];
    const float* WVw  = (const float*)d_in[7];
    const float* WVb  = (const float*)d_in[8];
    const float* Alog = (const float*)d_in[9];
    const float* Wg   = (const float*)d_in[10];

    char* wp = (char*)d_ws;
    auto carve = [&](size_t bytes) -> char* {
        char* p = wp;
        wp += (bytes + 255) & ~(size_t)255;
        return p;
    };
    unsigned short* xb  = (unsigned short*)carve((size_t)NTOK * DIMX * 2);
    unsigned short* wb4 = (unsigned short*)carve((size_t)4 * DIMX * DIMX * 2);
    unsigned short* wgb = (unsigned short*)carve((size_t)HEADS * HD * HD * 2);
    float* Bm = (float*)carve((size_t)NTOK * DIMX * 4);
    float* Cm = (float*)carve((size_t)NTOK * DIMX * 4);
    float* Vm = (float*)carve((size_t)NTOK * DIMX * 4);
    float* Am = (float*)carve((size_t)NTOK * DIMX * 4);
    unsigned short* CGb = (unsigned short*)carve((size_t)NBH * SEQ * HD * 2);
    unsigned short* BHb = (unsigned short*)carve((size_t)NBH * SEQ * HD * 2);
    unsigned short* Vtb = (unsigned short*)carve((size_t)NBH * HD * SEQ * 2);
    float* Diag = (float*)carve((size_t)NBH * SEQ * 4);

    // fp32 -> bf16 conversions
    int nx = NTOK * DIMX;
    cvt_bf16_kernel<<<(nx + 255) / 256, 256, 0, stream>>>(x, xb, nx);
    int nw = DIMX * DIMX;
    cvt_bf16_kernel<<<(nw + 255) / 256, 256, 0, stream>>>(WBw, wb4 + 0 * (size_t)nw, nw);
    cvt_bf16_kernel<<<(nw + 255) / 256, 256, 0, stream>>>(WCw, wb4 + 1 * (size_t)nw, nw);
    cvt_bf16_kernel<<<(nw + 255) / 256, 256, 0, stream>>>(WSw, wb4 + 2 * (size_t)nw, nw);
    cvt_bf16_kernel<<<(nw + 255) / 256, 256, 0, stream>>>(WVw, wb4 + 3 * (size_t)nw, nw);
    int ng = HEADS * HD * HD;
    cvt_bf16_kernel<<<(ng + 255) / 256, 256, 0, stream>>>(Wg, wgb, ng);

    // B/C/V/A projections (bf16 WMMA, f32 accumulate)
    proj_kernel<<<dim3(NTOK / 64, DIMX / 64, 4), 128, 0, stream>>>(
        xb, wb4, WBb, WCb, WSb, WVb, Alog, Bm, Cm, Am, Vm);

    // sequential per-channel gate scan
    scan_kernel<<<NBH, 64, 0, stream>>>(Bm, Cm, Vm, Am, CGb, BHb, Vtb, Diag);

    // causal rank-64 mixing + per-head channel mix
    attn_kernel<<<dim3(SEQ / 64, NBH), 128, 0, stream>>>(CGb, BHb, Vtb, Diag, wgb,
                                                         (float*)d_out);
}
